// FracDownSample_45414984188028
// MI455X (gfx1250) — matrix operand order repaired and checked by reading, not compiled, not run on hardware
//
#include <hip/hip_runtime.h>

typedef __attribute__((ext_vector_type(16))) __bf16 v16bf;
typedef __attribute__((ext_vector_type(8)))  float  v8f;
typedef __attribute__((ext_vector_type(4)))  int    v4i;
typedef __attribute__((ext_vector_type(8)))  int    v8i;

union BF16x16 { v16bf v; unsigned u[8]; uint4 q[2]; };

__device__ __forceinline__ unsigned pack_bf16(float lo, float hi) {
    unsigned ulo = __float_as_uint(lo);
    unsigned uhi = __float_as_uint(hi);
    ulo = (ulo + 0x7FFFu + ((ulo >> 16) & 1u)) >> 16;   // RNE
    uhi = (uhi + 0x7FFFu + ((uhi >> 16) & 1u)) >> 16;
    return ulo | (uhi << 16);
}

// Shapes: x(16,256,128,128) f32, w_key(64,256), w_sim(4,64); out (16,256,64,64)
// 32x32 windows of 4x4 px; workgroup = 8 waves = 8 wx-windows, 2 hy rows (pipelined)
#define XS_PITCH 130                       // 128 px + 2 pad dwords (TDM pad 6/1)
#define KQW_OFF  0                         // 512 floats: [8 waves][16 p][4 s]
#define WKB_OFF  512                       // 8192 dwords: packed bf16 B fragments
#define OUTS_OFF (512 + 8192)              // 4224 floats: [128 c][33] staging
#define XS0_OFF  (512 + 8192 + 4224)       // 33280 floats: slab 0
#define XS1_OFF  (XS0_OFF + 33280)         // 33280 floats: slab 1
#define SMEM_FLOATS (XS1_OFF + 33280)      // 79488 floats = 317952 B (< 320KB WGP)

// ---- Tensor Data Mover: 3D tile (32 x) x (4 rows, stride 128) x (256 c, stride 16384),
// ---- fp32, LDS padding 2 dwords per 128 dwords (=> per-channel pitch 130). ----
__device__ __forceinline__ void tdm_issue(const float* gtile, unsigned lds_byte) {
    unsigned long long ga = (unsigned long long)(uintptr_t)gtile;
    v4i g0 = { 1,                                   // count=1 (valid), user mode
               (int)lds_byte,                       // lds_addr (bytes)
               (int)(unsigned)(ga & 0xffffffffull), // global_addr[31:0]
               (int)((unsigned)((ga >> 32) & 0x01ffffffu) | 0x80000000u) }; // [56:32] | type=2
    v8i g1 = { (int)((2u << 16) | (1u << 20) | (6u << 22) | (1u << 25)),
               //  data_size=4B  pad_en      pad_int=128dw  pad_amt=2dw ; wg_mask=0
               (int)(128u << 16),                   // tensor_dim0 = 128 (low16 in [63:48])
               (int)(128u << 16),                   // dim0 hi=0 ; tensor_dim1 = 128
               (int)(32u << 16),                    // dim1 hi=0 ; tile_dim0 = 32
               (int)(4u | (256u << 16)),            // tile_dim1 = 4 ; tile_dim2 = 256
               128,                                 // tensor_dim0_stride = 128
               (int)(0x4000u << 16),                // stride0 hi=0 ; tensor_dim1_stride=16384 lo
               0 };                                 // stride1 hi = 0
    v4i g2 = { 256, 0, 0, 0 };                      // tensor_dim2=256; dim3/stride2/tile3 = 0
    v4i g3 = { 0, 0, 0, 0 };
    asm volatile("tensor_load_to_lds %0, %1, %2, %3"
                 :: "s"(g0), "s"(g1), "s"(g2), "s"(g3) : "memory");
}

extern "C" __global__ __launch_bounds__(256, 1)
void frac_down_sample_fused(const float* __restrict__ x,
                            const float* __restrict__ wkey,
                            const float* __restrict__ wsim,
                            float* __restrict__ out)
{
    extern __shared__ float smem[];
    float*    kqw  = smem + KQW_OFF;
    unsigned* wkB  = (unsigned*)(smem + WKB_OFF);
    float*    outs = smem + OUTS_OFF;

    const int t    = threadIdx.x;
    const int lane = t & 31;
    const int wv   = t >> 5;
    const int bid  = blockIdx.x;
    const int wx0  = (bid & 3) * 8;          // first window-x of slab
    const int hy0  = ((bid >> 2) & 15) * 2;  // first of two window-rows
    const int n    = bid >> 6;               // batch

    const float* xg = x + (size_t)n * 256 * 16384;

    // ---- kick off both slab DMAs immediately (wave 0; EXEC-independent op) ----
    if (wv == 0) {
        tdm_issue(xg + (size_t)(hy0 * 4) * 128 + wx0 * 4,
                  (unsigned)(uintptr_t)(void*)(smem + XS0_OFF));
        tdm_issue(xg + (size_t)((hy0 + 1) * 4) * 128 + wx0 * 4,
                  (unsigned)(uintptr_t)(void*)(smem + XS1_OFF));
    }

    // ---- overlap with DMA: repack w_key -> WMMA B fragments (bf16) in LDS ----
    // dword D = ((kc*4+nt)*32 + blane)*8 + vg ; B elem (k,ncol): blane = ncol+16*((k>>3)&1)
    for (int j = 0; j < 32; ++j) {
        int flat = j * 256 + t;              // 0..8191
        int vg   = flat & 7;
        int bl   = (flat >> 3) & 31;
        int nt   = (flat >> 8) & 3;
        int kc   = flat >> 10;
        int ncol = bl & 15, khi = bl >> 4;
        int ke   = 2 * (vg & 3) + 16 * (vg >> 2) + 8 * khi;
        int row  = nt * 16 + ncol;
        int ci   = kc * 32 + ke;
        wkB[flat] = pack_bf16(wkey[row * 256 + ci], wkey[row * 256 + ci + 1]);
    }

    // w_sim values this lane needs: wsr[s][nt] = w_sim[s, nt*16 + (lane&15)]
    const int p15  = lane & 15;
    const int hi16 = lane >> 4;
    float wsr[4][4];
    #pragma unroll
    for (int s = 0; s < 4; ++s)
        #pragma unroll
        for (int nt = 0; nt < 4; ++nt)
            wsr[s][nt] = wsim[s * 64 + nt * 16 + p15];

    if (wv == 0) __builtin_amdgcn_s_wait_tensorcnt(1);   // slab 0 landed
    __syncthreads();

    for (int it = 0; it < 2; ++it) {
        if (it == 1) {
            if (wv == 0) __builtin_amdgcn_s_wait_tensorcnt(0);  // slab 1 landed
            __syncthreads();
        }
        const float* xs = smem + (it ? XS1_OFF : XS0_OFF);
        const int hy = hy0 + it;

        // ---------- Phase 1: key GEMM via WMMA (M=16 px, K=256, N=64) ----------
        v8f zacc = {};
        v8f acc[4] = { zacc, zacc, zacc, zacc };
        const int xbase = wv * 16 + p15;

        for (int kc = 0; kc < 8; ++kc) {
            v16bf av;
            #pragma unroll
            for (int vi = 0; vi < 8; ++vi) {
                int ke = ((vi < 4) ? 0 : 16) + (hi16 ? 8 : 0) + 2 * (vi & 3);
                int c0 = kc * 32 + ke;
                float f0 = xs[c0 * XS_PITCH + xbase];
                float f1 = xs[(c0 + 1) * XS_PITCH + xbase];
                av[2 * vi]     = (__bf16)f0;
                av[2 * vi + 1] = (__bf16)f1;
            }
            #pragma unroll
            for (int nt = 0; nt < 4; ++nt) {
                BF16x16 bu;
                const uint4* bp = (const uint4*)(wkB + (((kc * 4 + nt) * 32 + lane) * 8));
                bu.q[0] = bp[0];
                bu.q[1] = bp[1];
                acc[nt] = __builtin_amdgcn_wmma_f32_16x16x32_bf16(
                    false, av, false, bu.v, (short)0, acc[nt], false, false);
            }
        }

        // ---------- Phase 2: kq = w_sim*relu(key); softmax over 16 px ----------
        float part[8][4];
        #pragma unroll
        for (int r = 0; r < 8; ++r)
            #pragma unroll
            for (int s = 0; s < 4; ++s) part[r][s] = 0.f;

        #pragma unroll
        for (int nt = 0; nt < 4; ++nt)
            #pragma unroll
            for (int r = 0; r < 8; ++r) {
                float kv = acc[nt][r];
                kv = kv > 0.f ? kv : 0.f;
                #pragma unroll
                for (int s = 0; s < 4; ++s)
                    part[r][s] = fmaf(wsr[s][nt], kv, part[r][s]);
            }

        #pragma unroll
        for (int r = 0; r < 8; ++r)
            #pragma unroll
            for (int s = 0; s < 4; ++s) {
                float v = part[r][s];
                v += __shfl_xor(v, 1);
                v += __shfl_xor(v, 2);
                v += __shfl_xor(v, 4);
                v += __shfl_xor(v, 8);
                part[r][s] = v * 0.25f;      // kq / si
            }

        float wgt[8][4];
        #pragma unroll
        for (int s = 0; s < 4; ++s) {
            float mx = part[0][s];
            #pragma unroll
            for (int r = 1; r < 8; ++r) mx = fmaxf(mx, part[r][s]);
            mx = fmaxf(mx, __shfl_xor(mx, 16));
            float sum = 0.f;
            #pragma unroll
            for (int r = 0; r < 8; ++r) {
                float e = __expf(part[r][s] - mx);
                wgt[r][s] = e;
                sum += e;
            }
            sum += __shfl_xor(sum, 16);
            float inv = 1.f / sum;
            #pragma unroll
            for (int r = 0; r < 8; ++r) wgt[r][s] *= inv;
        }

        if (p15 == 0) {                      // one lane per half-group publishes
            #pragma unroll
            for (int r = 0; r < 8; ++r) {
                int m = r + 8 * hi16;
                float4 w4 = { wgt[r][0], wgt[r][1], wgt[r][2], wgt[r][3] };
                *(float4*)(kqw + wv * 64 + m * 4) = w4;
            }
        }

        // hoist this window's 16x4 weights into registers once
        float4 kwv[16];
        #pragma unroll
        for (int p = 0; p < 16; ++p)
            kwv[p] = *(const float4*)(kqw + wv * 64 + p * 4);

        // ---------- Phase 3/4: aggregation + staged coalesced store ----------
        float* og = out + (size_t)n * 256 * 4096 + (size_t)(hy * 2) * 64 + wx0 * 2;
        for (int half = 0; half < 2; ++half) {
            #pragma unroll
            for (int i4 = 0; i4 < 4; ++i4) {
                int i  = half * 4 + i4;
                int c  = i * 32 + lane;
                int cl = c - half * 128;
                float xr[16];
                #pragma unroll
                for (int k = 0; k < 8; ++k)
                    *(float2*)&xr[2 * k] =
                        *(const float2*)(xs + c * XS_PITCH + wv * 16 + 2 * k);
                float o0 = 0.f, o1 = 0.f, o2 = 0.f, o3 = 0.f;
                #pragma unroll
                for (int p = 0; p < 16; ++p) {
                    o0 = fmaf(xr[p], kwv[p].x, o0);
                    o1 = fmaf(xr[p], kwv[p].y, o1);
                    o2 = fmaf(xr[p], kwv[p].z, o2);
                    o3 = fmaf(xr[p], kwv[p].w, o3);
                }
                outs[cl * 33 + 0 * 16 + wv * 2 + 0] = o0;  // q0 -> (qy0,qx0)
                outs[cl * 33 + 0 * 16 + wv * 2 + 1] = o1;  // q1 -> (0,1)
                outs[cl * 33 + 1 * 16 + wv * 2 + 0] = o2;  // q2 -> (1,0)
                outs[cl * 33 + 1 * 16 + wv * 2 + 1] = o3;  // q3 -> (1,1)
            }
            __syncthreads();
            #pragma unroll
            for (int j = 0; j < 4; ++j) {
                int flat = j * 256 + t;      // 0..1023 float4s
                int wq = flat & 3;
                int qy = (flat >> 2) & 1;
                int cl = flat >> 3;          // 0..127
                int c  = half * 128 + cl;
                const float* s = outs + cl * 33 + qy * 16 + wq * 4;
                float4 v = { s[0], s[1], s[2], s[3] };
                *(float4*)(og + (size_t)c * 4096 + qy * 64 + wq * 4) = v;
            }
            __syncthreads();                 // outs reused by next half/iteration
        }
    }
}

extern "C" void kernel_launch(void* const* d_in, const int* in_sizes, int n_in,
                              void* d_out, int out_size, void* d_ws, size_t ws_size,
                              hipStream_t stream) {
    (void)in_sizes; (void)n_in; (void)d_ws; (void)ws_size; (void)out_size;
    const float* x    = (const float*)d_in[0];
    const float* wkey = (const float*)d_in[1];
    const float* wsim = (const float*)d_in[2];
    float* out = (float*)d_out;
    // 16 batches * 16 hy-pairs * 4 wx-slabs = 1024 workgroups, 2 window-rows each
    dim3 grid(1024), block(256);
    size_t smem = (size_t)SMEM_FLOATS * sizeof(float);   // 317952 B
    frac_down_sample_fused<<<grid, block, smem, stream>>>(x, wkey, wsim, out);
}
